// MultiHeadAttention_19937238188090
// MI455X (gfx1250) — matrix operand order repaired and checked by reading, not compiled, not run on hardware
//
#include <hip/hip_runtime.h>

typedef __attribute__((ext_vector_type(16))) _Float16 v16h;
typedef __attribute__((ext_vector_type(8)))  _Float16 v8h;
typedef __attribute__((ext_vector_type(4)))  _Float16 v4h;
typedef __attribute__((ext_vector_type(8)))  float    v8f;

static constexpr int D_MODEL   = 1024;
static constexpr int NUM_HEADS = 16;
static constexpr int D_K       = 64;
static constexpr int BATCH     = 2;
static constexpr int SEQ       = 2048;

enum { STORE_PLAIN = 0, STORE_SPLIT_QK = 1, STORE_SPLIT_V = 2 };

// ---------------------------------------------------------------------------
// One-time fp32 -> (hi,lo) f16 split, vectorized x4. n must be a multiple of 4.
// ---------------------------------------------------------------------------
__global__ __launch_bounds__(256)
void split_pass(const float* __restrict__ x, _Float16* __restrict__ hi,
                _Float16* __restrict__ lo, long n)
{
    const long i = ((long)blockIdx.x * 256 + threadIdx.x) * 4;
    if (i >= n) return;
    const float4 v = *(const float4*)(x + i);
    v4h h, l;
    h[0] = (_Float16)v.x; l[0] = (_Float16)(v.x - (float)h[0]);
    h[1] = (_Float16)v.y; l[1] = (_Float16)(v.y - (float)h[1]);
    h[2] = (_Float16)v.z; l[2] = (_Float16)(v.z - (float)h[2]);
    h[3] = (_Float16)v.w; l[3] = (_Float16)(v.w - (float)h[3]);
    *(v4h*)(hi + i) = h;
    *(v4h*)(lo + i) = l;
}

// ---------------------------------------------------------------------------
// C[z][m,n] = alpha * sum_k A[z][m,k] * Bt[z][n,k]  (+ bias[n])
// Operands are pre-split f16 hi/lo (or, for A_IS_F32, A is fp32 and split
// in-kernel — used only where each element feeds exactly one wave tile).
// Error-compensated product: aHi*bHi + aHi*bLo + aLo*bHi, fp32 accumulators.
// Wave tile: 32(M) x 64(N) -> 8 accumulators, 24 WMMA per k-step against
// ~16 vector loads. Outputs can be written as fp32 (C) and/or f16 hi/lo
// pairs (Chi/Clo) for the next consumer.
// ---------------------------------------------------------------------------
template <int MODE, bool A_IS_F32>
__global__ __launch_bounds__(256)
void gemm_wmma(const float* __restrict__ Af32,
               const _Float16* __restrict__ Ahi, const _Float16* __restrict__ Alo,
               int lda, long aBatch,
               const _Float16* __restrict__ Bhi, const _Float16* __restrict__ Blo,
               int ldb, long bBatch,
               const float* __restrict__ bias,
               float* __restrict__ C,
               _Float16* __restrict__ Chi, _Float16* __restrict__ Clo,
               int ldc, int cDiv, long cOuter, long cInner,
               int M, int N, int K, float alpha)
{
    const int lane = threadIdx.x & 31;
    const int wave = threadIdx.x >> 5;
    const int half = lane >> 4;        // 0: lanes 0-15, 1: lanes 16-31
    const int l16  = lane & 15;

    const int mBase = (blockIdx.y * 8 + wave) * 32;   // two 16-row tiles
    const int nBase = blockIdx.x * 64;
    const int z     = blockIdx.z;
    if (mBase >= M) return;

    const v8f vzero = {0.f, 0.f, 0.f, 0.f, 0.f, 0.f, 0.f, 0.f};
    v8f acc[2][4];
#pragma unroll
    for (int r = 0; r < 2; ++r)
#pragma unroll
        for (int t = 0; t < 4; ++t) acc[r][t] = vzero;

    for (int k0 = 0; k0 < K; k0 += 32) {
        // ---- A fragments (16x32 each): lane m = l%16,
        //      K = 16*half + e
        v16h aHi[2], aLo[2];
#pragma unroll
        for (int r = 0; r < 2; ++r) {
            const long rowOff = (long)(mBase + r * 16 + l16) * lda;
            if (A_IS_F32) {
                const float* Arow = Af32 + (long)z * aBatch + rowOff;
#pragma unroll
                for (int e = 0; e < 8; ++e) {
                    float x0 = Arow[k0 + half * 16 + e];
                    _Float16 h0 = (_Float16)x0;
                    aHi[r][e]     = h0;
                    aLo[r][e]     = (_Float16)(x0 - (float)h0);
                    float x1 = Arow[k0 + 8 + half * 16 + e];
                    _Float16 h1 = (_Float16)x1;
                    aHi[r][e + 8] = h1;
                    aLo[r][e + 8] = (_Float16)(x1 - (float)h1);
                }
            } else {
                const _Float16* ah = Ahi + (long)z * aBatch + rowOff;
                const _Float16* al = Alo + (long)z * aBatch + rowOff;
                v8h h0 = *(const v8h*)(ah + k0 + half * 16);
                v8h h1 = *(const v8h*)(ah + k0 + 8 + half * 16);
                v8h l0 = *(const v8h*)(al + k0 + half * 16);
                v8h l1 = *(const v8h*)(al + k0 + 8 + half * 16);
                aHi[r] = __builtin_shufflevector(h0, h1, 0,1,2,3,4,5,6,7,8,9,10,11,12,13,14,15);
                aLo[r] = __builtin_shufflevector(l0, l1, 0,1,2,3,4,5,6,7,8,9,10,11,12,13,14,15);
            }
        }

#pragma unroll
        for (int t = 0; t < 4; ++t) {
            // ---- B fragment (32x16): lane n = l%16, K = 16*half + e
            const long rowOff = (long)(nBase + t * 16 + l16) * ldb;
            const v16h bHi = *(const v16h*)(Bhi + (long)z * bBatch + rowOff + k0 + half * 16);
            const v16h bLo = *(const v16h*)(Blo + (long)z * bBatch + rowOff + k0 + half * 16);
#pragma unroll
            for (int r = 0; r < 2; ++r) {
                acc[r][t] = __builtin_amdgcn_wmma_f32_16x16x32_f16(
                    false, aHi[r], false, bLo, (short)0, acc[r][t], false, false);
                acc[r][t] = __builtin_amdgcn_wmma_f32_16x16x32_f16(
                    false, aLo[r], false, bHi, (short)0, acc[r][t], false, false);
                acc[r][t] = __builtin_amdgcn_wmma_f32_16x16x32_f16(
                    false, aHi[r], false, bHi, (short)0, acc[r][t], false, false);
            }
        }
    }

    // ---- store (C layout: m = i + 8*half, n = l%16)
#pragma unroll
    for (int r = 0; r < 2; ++r) {
#pragma unroll
        for (int t = 0; t < 4; ++t) {
#pragma unroll
            for (int i = 0; i < 8; ++i) {
                const int m = mBase + r * 16 + i + 8 * half;
                const int n = nBase + t * 16 + l16;
                float v = acc[r][t][i] * alpha;
                if (bias) v += bias[n];
                long idx;
                if (MODE == STORE_PLAIN) {
                    const long base = (long)(z / cDiv) * cOuter + (long)(z % cDiv) * cInner;
                    idx = base + (long)m * ldc + n;
                } else if (MODE == STORE_SPLIT_QK) {
                    // [B*S, D] -> [B, H, S, Dk]
                    const int b = m / SEQ, s = m % SEQ, h = n / D_K, d = n % D_K;
                    idx = (((long)(b * NUM_HEADS + h) * SEQ) + s) * D_K + d;
                } else { // STORE_SPLIT_V: -> [B, H, Dk, S]  (V transposed)
                    const int b = m / SEQ, s = m % SEQ, h = n / D_K, d = n % D_K;
                    idx = (((long)(b * NUM_HEADS + h) * D_K) + d) * SEQ + s;
                }
                if (C) C[idx] = v;
                if (Chi) {
                    _Float16 h16 = (_Float16)v;
                    Chi[idx] = h16;
                    Clo[idx] = (_Float16)(v - (float)h16);
                }
            }
        }
    }
}

// ---------------------------------------------------------------------------
// In-place row softmax over the attn region (rows of length SEQ=2048).
// Register-cached: one read + one write pass per row.
// ---------------------------------------------------------------------------
__global__ __launch_bounds__(256)
void softmax_rows(float* __restrict__ attn)
{
    const int tid = threadIdx.x;
    float* row = attn + (long)blockIdx.x * SEQ;

    float v[8];
    float m = -3.402823466e38f;
#pragma unroll
    for (int j = 0; j < 8; ++j) {
        v[j] = row[tid + j * 256];
        m = fmaxf(m, v[j]);
    }
#pragma unroll
    for (int off = 16; off > 0; off >>= 1)
        m = fmaxf(m, __shfl_down(m, off, 32));

    __shared__ float smax[8];
    __shared__ float ssum[8];
    if ((tid & 31) == 0) smax[tid >> 5] = m;
    __syncthreads();
    m = smax[0];
#pragma unroll
    for (int w = 1; w < 8; ++w) m = fmaxf(m, smax[w]);

    float sum = 0.f;
#pragma unroll
    for (int j = 0; j < 8; ++j) {
        v[j] = expf(v[j] - m);
        sum += v[j];
    }
#pragma unroll
    for (int off = 16; off > 0; off >>= 1)
        sum += __shfl_down(sum, off, 32);
    if ((tid & 31) == 0) ssum[tid >> 5] = sum;
    __syncthreads();
    sum = 0.f;
#pragma unroll
    for (int w = 0; w < 8; ++w) sum += ssum[w];

    const float inv = 1.0f / sum;
#pragma unroll
    for (int j = 0; j < 8; ++j)
        row[tid + j * 256] = v[j] * inv;
}

// ---------------------------------------------------------------------------
extern "C" void kernel_launch(void* const* d_in, const int* in_sizes, int n_in,
                              void* d_out, int out_size, void* d_ws, size_t ws_size,
                              hipStream_t stream)
{
    const float* q   = (const float*)d_in[0];
    const float* k   = (const float*)d_in[1];
    const float* v   = (const float*)d_in[2];
    const float* w_q = (const float*)d_in[3];
    const float* b_q = (const float*)d_in[4];
    const float* w_k = (const float*)d_in[5];
    const float* b_k = (const float*)d_in[6];
    const float* w_v = (const float*)d_in[7];
    const float* b_v = (const float*)d_in[8];
    const float* w_o = (const float*)d_in[9];
    const float* b_o = (const float*)d_in[10];

    float* out  = (float*)d_out;                              // [B, S, D]
    float* attn = out + (long)BATCH * SEQ * D_MODEL;          // [B, H, S, S]

    const long NACT = (long)BATCH * SEQ * D_MODEL;            // 4 Mi elements
    const long NW   = (long)D_MODEL * D_MODEL;                // 1 Mi elements

    _Float16* p = (_Float16*)d_ws;
    _Float16* Qhi  = p; p += NACT;   // [B,H,S,Dk]
    _Float16* Qlo  = p; p += NACT;
    _Float16* Khi  = p; p += NACT;   // [B,H,S,Dk]
    _Float16* Klo  = p; p += NACT;
    _Float16* Vthi = p; p += NACT;   // [B,H,Dk,S]
    _Float16* Vtlo = p; p += NACT;
    _Float16* Hohi = p; p += NACT;   // [B*S, D]
    _Float16* Holo = p; p += NACT;
    _Float16* Xhi  = p; p += NACT;   // activation split scratch (reused)
    _Float16* Xlo  = p; p += NACT;
    _Float16* Whi  = p; p += NW;     // weight split scratch (reused)
    _Float16* Wlo  = p; p += NW;

    const dim3 blk(256, 1, 1);
    const int MROWS = BATCH * SEQ;                            // 4096
    const dim3 gProj(D_MODEL / 64, MROWS / 256, 1);
    const dim3 gSplitA((unsigned)((NACT / 4 + 255) / 256), 1, 1);
    const dim3 gSplitW((unsigned)((NW / 4 + 255) / 256), 1, 1);

    // ---- Q projection
    split_pass<<<gSplitA, blk, 0, stream>>>(q, Xhi, Xlo, NACT);
    split_pass<<<gSplitW, blk, 0, stream>>>(w_q, Whi, Wlo, NW);
    gemm_wmma<STORE_SPLIT_QK, false><<<gProj, blk, 0, stream>>>(
        nullptr, Xhi, Xlo, D_MODEL, 0, Whi, Wlo, D_MODEL, 0, b_q,
        nullptr, Qhi, Qlo, 0, 1, 0, 0, MROWS, D_MODEL, D_MODEL, 1.0f);

    // ---- K projection
    split_pass<<<gSplitA, blk, 0, stream>>>(k, Xhi, Xlo, NACT);
    split_pass<<<gSplitW, blk, 0, stream>>>(w_k, Whi, Wlo, NW);
    gemm_wmma<STORE_SPLIT_QK, false><<<gProj, blk, 0, stream>>>(
        nullptr, Xhi, Xlo, D_MODEL, 0, Whi, Wlo, D_MODEL, 0, b_k,
        nullptr, Khi, Klo, 0, 1, 0, 0, MROWS, D_MODEL, D_MODEL, 1.0f);

    // ---- V projection (written transposed per head)
    split_pass<<<gSplitA, blk, 0, stream>>>(v, Xhi, Xlo, NACT);
    split_pass<<<gSplitW, blk, 0, stream>>>(w_v, Whi, Wlo, NW);
    gemm_wmma<STORE_SPLIT_V, false><<<gProj, blk, 0, stream>>>(
        nullptr, Xhi, Xlo, D_MODEL, 0, Whi, Wlo, D_MODEL, 0, b_v,
        nullptr, Vthi, Vtlo, 0, 1, 0, 0, MROWS, D_MODEL, D_MODEL, 1.0f);

    // ---- scores = (Q @ K^T) / sqrt(Dk), raw into the attn output region
    const dim3 gScore(SEQ / 64, SEQ / 256, BATCH * NUM_HEADS);
    gemm_wmma<STORE_PLAIN, false><<<gScore, blk, 0, stream>>>(
        nullptr, Qhi, Qlo, D_K, (long)SEQ * D_K, Khi, Klo, D_K, (long)SEQ * D_K,
        nullptr, attn, nullptr, nullptr,
        SEQ, 1, (long)SEQ * SEQ, 0, SEQ, SEQ, D_K, 0.125f);

    // ---- softmax in place
    softmax_rows<<<dim3(BATCH * NUM_HEADS * SEQ, 1, 1), blk, 0, stream>>>(attn);

    // ---- head_out = attn @ V (A = fp32 attn split in-kernel; B = Vt f16),
    //      merged to [B*S, D] as f16 hi/lo: base = b*S*D + h*Dk
    const dim3 gAV(D_K / 64, SEQ / 256, BATCH * NUM_HEADS);
    gemm_wmma<STORE_PLAIN, true><<<gAV, blk, 0, stream>>>(
        attn, nullptr, nullptr, SEQ, (long)SEQ * SEQ, Vthi, Vtlo, SEQ, (long)D_K * SEQ,
        nullptr, nullptr, Hohi, Holo,
        D_MODEL, NUM_HEADS, (long)SEQ * D_MODEL, D_K, SEQ, D_K, SEQ, 1.0f);

    // ---- final projection: out = head_out @ w_o^T + b_o
    split_pass<<<gSplitW, blk, 0, stream>>>(w_o, Whi, Wlo, NW);
    gemm_wmma<STORE_PLAIN, false><<<gProj, blk, 0, stream>>>(
        nullptr, Hohi, Holo, D_MODEL, 0, Whi, Wlo, D_MODEL, 0, b_o,
        out, nullptr, nullptr, D_MODEL, 1, 0, 0, MROWS, D_MODEL, D_MODEL, 1.0f);
}
